// GCNLayer_67654324846792
// MI455X (gfx1250) — matrix operand order repaired and checked by reading, not compiled, not run on hardware
//
#include <hip/hip_runtime.h>
#include <hip/hip_bf16.h>

#define NN 100000
#define NE 3200000
#define F  128

typedef __attribute__((ext_vector_type(16))) __bf16 v16bf;
typedef __attribute__((ext_vector_type(8)))  float  v8f;
typedef __attribute__((ext_vector_type(4)))  float  f32x4;

__device__ __forceinline__ v8f wmma_bf16(v16bf a, v16bf b, v8f c) {
    // D = A(16x32 bf16) * B(32x16 bf16) + C(16x16 f32)
    return __builtin_amdgcn_wmma_f32_16x16x32_bf16(
        /*neg_a=*/false, a, /*neg_b=*/false, b,
        /*c_mod=*/(short)0, c, /*reuse_a=*/false, /*reuse_b=*/false);
}

// Split 4 fp32 values into bf16 hi + bf16 residual lo, writing slots [base, base+4)
__device__ __forceinline__ void split4(v16bf& hi, v16bf& lo, int base, f32x4 v) {
    float f0 = v.x, f1 = v.y, f2 = v.z, f3 = v.w;
    __bf16 h0 = (__bf16)f0, h1 = (__bf16)f1, h2 = (__bf16)f2, h3 = (__bf16)f3;
    hi[base + 0] = h0; lo[base + 0] = (__bf16)(f0 - (float)h0);
    hi[base + 1] = h1; lo[base + 1] = (__bf16)(f1 - (float)h1);
    hi[base + 2] = h2; lo[base + 2] = (__bf16)(f2 - (float)h2);
    hi[base + 3] = h3; lo[base + 3] = (__bf16)(f3 - (float)h3);
}

// ---------------- degree kernels ----------------

__global__ void __launch_bounds__(256) k_deg_init(float* __restrict__ deg) {
    int i = blockIdx.x * 256 + threadIdx.x;
    if (i < NN) deg[i] = 1.0f;          // self-loop contributes 1 to every node
}

__global__ void __launch_bounds__(256) k_deg_count(const long long* __restrict__ ei,
                                                   float* __restrict__ deg) {
    int e = blockIdx.x * 256 + threadIdx.x;
    if (e < NE) unsafeAtomicAdd(&deg[(int)ei[e]], 1.0f);   // row = ei[0][e]
}

__global__ void __launch_bounds__(256) k_rsqrt(float* __restrict__ deg) {
    int i = blockIdx.x * 256 + threadIdx.x;
    if (i < NN) deg[i] = rsqrtf(deg[i]);   // deg >= 1 always (self loop)
}

// ---------------- WMMA GEMM: support = x @ W ; out = bias + dinv^2 * support ----

__global__ void __launch_bounds__(128) k_gemm(const float* __restrict__ x,
                                              const float* __restrict__ w,
                                              const float* __restrict__ bias,
                                              const float* __restrict__ dinv,
                                              float* __restrict__ support,
                                              float* __restrict__ out) {
    const int lane  = threadIdx.x & 31;
    const int wave  = threadIdx.x >> 5;
    const int mtile = blockIdx.x * 4 + wave;
    if (mtile >= NN / 16) return;
    const int m0   = mtile * 16;
    const int mrow = m0 + (lane & 15);        // A: lane holds row m = lane&15
    const int kgrp = (lane >> 4) * 8;         // A: k sub-group per lane half

    v8f acc[8];
    #pragma unroll
    for (int n = 0; n < 8; ++n)
        #pragma unroll
        for (int j = 0; j < 8; ++j) acc[n][j] = 0.0f;

    #pragma unroll
    for (int ks = 0; ks < 4; ++ks) {          // K = 128 in 4 steps of 32
        const int k0 = ks * 32;

        // ---- A fragment (16x32 bf16): k = k0 + (i>>3)*16 + kgrp + (i&7)
        const float* xp = x + (size_t)mrow * F + k0 + kgrp;
        v16bf a_hi, a_lo;
        split4(a_hi, a_lo,  0, *(const f32x4*)(xp + 0));
        split4(a_hi, a_lo,  4, *(const f32x4*)(xp + 4));
        split4(a_hi, a_lo,  8, *(const f32x4*)(xp + 16));
        split4(a_hi, a_lo, 12, *(const f32x4*)(xp + 20));

        // ---- B fragments: lane holds W column n = n*16 + (lane&15), k as A
        const float* wp = w + (size_t)(k0 + kgrp) * F + (lane & 15);
        #pragma unroll
        for (int n = 0; n < 8; ++n) {
            v16bf b_hi, b_lo;
            split4(b_hi, b_lo,  0, (f32x4){wp[n * 16 + 0 * F], wp[n * 16 + 1 * F], wp[n * 16 + 2 * F], wp[n * 16 + 3 * F]});
            split4(b_hi, b_lo,  4, (f32x4){wp[n * 16 + 4 * F], wp[n * 16 + 5 * F], wp[n * 16 + 6 * F], wp[n * 16 + 7 * F]});
            split4(b_hi, b_lo,  8, (f32x4){wp[n * 16 + 16 * F], wp[n * 16 + 17 * F], wp[n * 16 + 18 * F], wp[n * 16 + 19 * F]});
            split4(b_hi, b_lo, 12, (f32x4){wp[n * 16 + 20 * F], wp[n * 16 + 21 * F], wp[n * 16 + 22 * F], wp[n * 16 + 23 * F]});
            // 3-term bf16 split ~= fp32 GEMM
            acc[n] = wmma_bf16(a_hi, b_hi, acc[n]);
            acc[n] = wmma_bf16(a_hi, b_lo, acc[n]);
            acc[n] = wmma_bf16(a_lo, b_hi, acc[n]);
        }
    }

    // ---- epilogue: D element j -> row 8*(lane>>4)+j, col lane&15
    const int nlane = lane & 15;
    const int mbase = (lane >> 4) * 8;
    #pragma unroll
    for (int j = 0; j < 8; ++j) {
        const int m = m0 + mbase + j;
        const float dm = dinv[m];
        const float self_w = dm * dm;          // self-loop norm = dinv[m]^2
        #pragma unroll
        for (int n = 0; n < 8; ++n) {
            const int nn = n * 16 + nlane;
            const float v = acc[n][j];
            support[(size_t)m * F + nn] = v;
            out[(size_t)m * F + nn] = bias[nn] + self_w * v;
        }
    }
}

// ---------------- edge scatter: out[row] += dinv[row]*dinv[col] * support[col] --

__global__ void __launch_bounds__(256) k_scatter(const long long* __restrict__ ei,
                                                 const float* __restrict__ dinv,
                                                 const float* __restrict__ support,
                                                 float* __restrict__ out) {
    const int lane = threadIdx.x & 31;
    const int e = blockIdx.x * 8 + (threadIdx.x >> 5);   // one wave32 per edge
    if (e >= NE) return;
    const int r = (int)ei[e];         // destination (row)
    const int c = (int)ei[NE + e];    // source (col)
    const float wgt = dinv[r] * dinv[c];
    // 51.2 MB support + 51.2 MB out both live in the 192 MB L2: random
    // gather + f32 atomics stay L2-resident.
    const f32x4 s = *(const f32x4*)(support + (size_t)c * F + lane * 4);
    float* op = out + (size_t)r * F + lane * 4;
    unsafeAtomicAdd(op + 0, wgt * s.x);
    unsafeAtomicAdd(op + 1, wgt * s.y);
    unsafeAtomicAdd(op + 2, wgt * s.z);
    unsafeAtomicAdd(op + 3, wgt * s.w);
}

// ---------------- launch ----------------

extern "C" void kernel_launch(void* const* d_in, const int* in_sizes, int n_in,
                              void* d_out, int out_size, void* d_ws, size_t ws_size,
                              hipStream_t stream) {
    const float*     x    = (const float*)d_in[0];
    const long long* ei   = (const long long*)d_in[1];   // int64 [2, NE]
    const float*     w    = (const float*)d_in[2];
    const float*     bias = (const float*)d_in[3];
    float* out = (float*)d_out;

    float* deg     = (float*)d_ws;                        // NN floats (becomes dinv)
    float* support = (float*)((char*)d_ws + 400384);      // NN*F floats, 128B-aligned

    k_deg_init <<<(NN + 255) / 256, 256, 0, stream>>>(deg);
    k_deg_count<<<(NE + 255) / 256, 256, 0, stream>>>(ei, deg);
    k_rsqrt    <<<(NN + 255) / 256, 256, 0, stream>>>(deg);

    const int mtiles = NN / 16;                           // 6250, exact
    k_gemm<<<(mtiles + 3) / 4, 128, 0, stream>>>(x, w, bias, deg, support, out);

    k_scatter<<<NE / 8, 256, 0, stream>>>(ei, deg, support, out);
}
